// DCTiDCTWrapper3D_44238163149474
// MI455X (gfx1250) — compile-verified, matrix operand order checked by reading, not run on hardware
//
#include <hip/hip_runtime.h>
#include <hip/hip_bf16.h>
#include <math.h>

// ---------------------------------------------------------------------------
// Shapes (fixed by the reference): B=16, T=8, C=1024, H=W=14
//   M  = B*T*H*W = 25088   (positions, channels-last rows)
//   C  = 1024              (channel dim; GEMM K / N)
//   NB = 576               (rfft bins 513 padded to 576 = 18*32 for tiling)
// Pipeline:
//   Xc   = transpose(x)                      [M, C]   bf16
//   Re,Im= (Xc @ Frt^T) .* wfilt             [M, NB]  bf16  (complex filter in epilogue)
//   Dm   = Re @ Brt^T + Im @ Bit^T           [M, C]   bf16  (irfft)
//   out  = x + gelu(Dm @ conv_w^T + bias)    scattered back to [bt, c, h, w]
// GEMM tile staging: Tensor Data Mover (tensor_load_to_lds, TENSORcnt),
// double-buffered LDS, one TDM descriptor per wave per k-step.
// ---------------------------------------------------------------------------

typedef __attribute__((ext_vector_type(16))) __bf16 v16bf;
typedef __attribute__((ext_vector_type(8)))  __bf16 v8bf;
typedef __attribute__((ext_vector_type(8)))  float  v8f;
typedef __attribute__((ext_vector_type(4)))  unsigned u32x4;
typedef __attribute__((ext_vector_type(8)))  unsigned u32x8;

#define M_ROWS   25088
#define CDIM     1024
#define NBIN     576      // padded bin count (valid: 0..512)
#define HWDIM    196
#define LDA      40       // LDS row stride in bf16 (80B: 16B aligned, bank-skewed)

// ---------------- CDNA5 WMMA / TDM / async helpers ----------------

__device__ __forceinline__ v8f wmma_bf16(v16bf a, v16bf b, v8f c) {
  // D = A(16x32 bf16) x B(32x16 bf16) + C(16x16 f32)
  return __builtin_amdgcn_wmma_f32_16x16x32_bf16(
      /*neg_a=*/false, a, /*neg_b=*/false, b,
      /*c_mod=*/(short)0, c, /*reuse_a=*/false, /*reuse_b=*/false);
}

// TDM: DMA one [rows x 32] bf16 tile (global row stride = stride_elems elements)
// into LDS with 16B padding after every 64B row => LDS row stride = 80B = LDA.
//   pad_interval=3 (16 DWORDs of data), pad_amount=3 (4 DWORDs of pad).
// Descriptor dwords are wave-uniform; readfirstlane pins them for "s" operands.
__device__ __forceinline__ void tdm_load_tile(void* lds_ptr, const void* gptr,
                                              unsigned rows, unsigned stride_elems) {
  unsigned la  = __builtin_amdgcn_readfirstlane((unsigned)(uintptr_t)lds_ptr);
  unsigned long long ga = (unsigned long long)(uintptr_t)gptr;
  unsigned glo = __builtin_amdgcn_readfirstlane((unsigned)ga);
  unsigned ghi = __builtin_amdgcn_readfirstlane((unsigned)(ga >> 32));
  unsigned rw  = __builtin_amdgcn_readfirstlane(rows);

  u32x4 g0;
  g0.x = 1u;                                  // count=1, user mode, no gather
  g0.y = la;                                  // lds_addr (low 32 bits of generic)
  g0.z = glo;                                 // global_addr[31:0]
  g0.w = (ghi & 0x01ffffffu) | (2u << 30);    // global_addr[56:32], type=2

  u32x8 g1;
  g1.s0 = (1u << 16)                          // data_size = 2 bytes
        | (1u << 20)                          // pad_enable
        | (3u << 22)                          // pad_interval: 16 DWORDs
        | (3u << 25);                         // pad_amount:   4 DWORDs
  g1.s1 = (stride_elems & 0xffffu) << 16;     // tensor_dim0 = stride (lo16)
  g1.s2 = (stride_elems >> 16) & 0xffffu;     // tensor_dim0 hi | tensor_dim1 lo16=0
  g1.s3 = 0x10u | (32u << 16);                // tensor_dim1 = 0x100000 | tile_dim0=32
  g1.s4 = rw;                                 // tile_dim1 = rows, tile_dim2 = 0
  g1.s5 = stride_elems;                       // tensor_dim0_stride (lo32)
  g1.s6 = 0u;
  g1.s7 = 0u;

  asm volatile("tensor_load_to_lds %0, %1" :: "s"(g0), "s"(g1) : "memory");
}

__device__ __forceinline__ void wait_tensor0() {
  __builtin_amdgcn_s_wait_tensorcnt(0);
}

// Per-lane async DMA: 4B global -> LDS (used by the f32 transpose stage).
__device__ __forceinline__ void async_load_b32(void* lds_ptr, const void* gptr) {
  unsigned lds_addr = (unsigned)(uintptr_t)lds_ptr;
  asm volatile("global_load_async_to_lds_b32 %0, %1, off"
               :: "v"(lds_addr), "v"(gptr)
               : "memory");
}

__device__ __forceinline__ void wait_async0() {
  asm volatile("s_wait_asynccnt 0x0" ::: "memory");
}

// Load one 16x32 bf16 fragment from an LDS tile (row-major, stride LDA).
// Per ISA layout: lanes 0-15 hold row r=lane, K={0..7} and {16..23};
// lanes 16-31 hold row r=lane-16, K={8..15} and {24..31}.
__device__ __forceinline__ v16bf lds_frag(const __bf16* base, int lane) {
  int r = lane & 15;
  int h = lane >> 4;                       // 0 or 1
  const __bf16* p = base + r * LDA + h * 8;
  v8bf lo = *(const v8bf*)(p);             // K = 8h .. 8h+7
  v8bf hi = *(const v8bf*)(p + 16);        // K = 16+8h .. 16+8h+7
  return __builtin_shufflevector(lo, hi, 0,1,2,3,4,5,6,7,8,9,10,11,12,13,14,15);
}

// ---------------- setup kernels ----------------

// Forward rDFT basis Frt/Fit: [NBIN, CDIM], bin-major (K-contiguous for GEMM B).
// Inverse basis Brt/Bit:      [CDIM, NBIN], channel-major (K=bin contiguous).
// Ortho norm: 1/sqrt(1024) = 1/32. Inverse uses half-spectrum weights (1,2,...,2,1).
__global__ __launch_bounds__(256)
void init_basis_kernel(__bf16* __restrict__ Frt, __bf16* __restrict__ Fit,
                       __bf16* __restrict__ Brt, __bf16* __restrict__ Bit) {
  int id = blockIdx.x * 256 + threadIdx.x;
  if (id >= NBIN * CDIM) return;
  int bin = id >> 10;
  int c   = id & 1023;
  int p   = (bin * c) & 1023;                      // exact phase mod N
  float ang = (float)p * (6.28318530717958647692f / 1024.0f);
  float cs = __cosf(ang), sn = __sinf(ang);
  float fs = (bin < 513) ? (1.0f / 32.0f) : 0.0f;
  Frt[id] = (__bf16)( cs * fs);
  Fit[id] = (__bf16)(-sn * fs);
  float w  = (bin == 0 || bin == 512) ? 1.0f : 2.0f;
  float is = (bin < 513) ? (w / 32.0f) : 0.0f;
  size_t io = (size_t)c * NBIN + bin;
  Brt[io] = (__bf16)( cs * is);
  Bit[io] = (__bf16)(-sn * is);
}

__global__ __launch_bounds__(256)
void convert_w_kernel(const float* __restrict__ conv_w, __bf16* __restrict__ Wb) {
  int id = blockIdx.x * 256 + threadIdx.x;
  if (id < CDIM * CDIM) Wb[id] = (__bf16)conv_w[id];
}

// x [bt, c, h*w] f32 -> Xc [bt*hw, c] bf16 (channels-last).
// Loads stage through async-to-LDS; s-clamped lanes land in tile slots that the
// store side never reads (store is guarded on s).
__global__ __launch_bounds__(256)
void transpose_cl_kernel(const float* __restrict__ x, __bf16* __restrict__ Xc) {
  __shared__ float tile[32][33];
  int bt = blockIdx.x;
  int s0 = blockIdx.y * 32;
  int c0 = blockIdx.z * 32;
  int tx = threadIdx.x, ty = threadIdx.y;
#pragma unroll
  for (int i = 0; i < 4; ++i) {
    int c = c0 + ty + i * 8;
    int s = s0 + tx;
    int sc = (s < HWDIM) ? s : (HWDIM - 1);     // clamp: padded slots never read
    async_load_b32(&tile[ty + i * 8][tx], &x[((size_t)bt * CDIM + c) * HWDIM + sc]);
  }
  wait_async0();
  __syncthreads();
#pragma unroll
  for (int i = 0; i < 4; ++i) {
    int s = s0 + ty + i * 8;
    int c = c0 + tx;
    if (s < HWDIM)
      Xc[((size_t)bt * HWDIM + s) * CDIM + c] = (__bf16)tile[tx][ty + i * 8];
  }
}

// ---------------- GEMM 1: forward rDFT + complex spectral filter ----------------
// Re/Im[m, bin] = sum_c Xc[m,c] * {Frt,Fit}[bin,c], then (Re+iIm) *= wfilt(t,hw,bin)

__global__ __launch_bounds__(128)
void gemm_fwd_fft_kernel(const __bf16* __restrict__ Xc,
                         const __bf16* __restrict__ Frt,
                         const __bf16* __restrict__ Fit,
                         const float*  __restrict__ wfilt,   // [8,196,513,2]
                         __bf16* __restrict__ Re, __bf16* __restrict__ Im) {
  __shared__ __bf16 sA[2][64 * LDA];
  __shared__ __bf16 sBr[2][64 * LDA];
  __shared__ __bf16 sBi[2][64 * LDA];
  int tid  = threadIdx.x;
  int lane = tid & 31;
  int wave = tid >> 5;                 // 4 waves
  int wm = wave >> 1, wn = wave & 1;   // 2x2 wave grid, 32x32 per wave
  int m0 = blockIdx.x * 64;
  int n0 = blockIdx.y * 64;

  // Per-wave TDM job: waves 0/3 split the A tile, waves 1/2 take Br/Bi.
  const __bf16* gbase;
  __bf16 *l0, *l1;
  unsigned rows;
  if (wave == 0)      { gbase = Xc  + (size_t)m0 * CDIM;        l0 = sA[0];            l1 = sA[1];            rows = 32; }
  else if (wave == 1) { gbase = Frt + (size_t)n0 * CDIM;        l0 = sBr[0];           l1 = sBr[1];           rows = 64; }
  else if (wave == 2) { gbase = Fit + (size_t)n0 * CDIM;        l0 = sBi[0];           l1 = sBi[1];           rows = 64; }
  else                { gbase = Xc  + (size_t)(m0 + 32) * CDIM; l0 = sA[0] + 32 * LDA; l1 = sA[1] + 32 * LDA; rows = 32; }

  tdm_load_tile(l0, gbase, rows, CDIM);

  v8f accR[4] = {};
  v8f accI[4] = {};

  for (int k0 = 0; k0 < CDIM; k0 += 32) {
    int cur = (k0 >> 5) & 1;
    wait_tensor0();                 // my tile for buffer `cur` landed
    __syncthreads();                // everyone's tiles landed; prev reads drained
    if (k0 + 32 < CDIM)
      tdm_load_tile(cur ? l0 : l1, gbase + k0 + 32, rows, CDIM);  // overlap DMA

    const __bf16* aT  = sA[cur];
    const __bf16* brT = sBr[cur];
    const __bf16* biT = sBi[cur];
    v16bf a[2], br[2], bi[2];
#pragma unroll
    for (int mt = 0; mt < 2; ++mt) a[mt]  = lds_frag(aT  + (wm * 32 + mt * 16) * LDA, lane);
#pragma unroll
    for (int nt = 0; nt < 2; ++nt) { br[nt] = lds_frag(brT + (wn * 32 + nt * 16) * LDA, lane);
                                     bi[nt] = lds_frag(biT + (wn * 32 + nt * 16) * LDA, lane); }
#pragma unroll
    for (int mt = 0; mt < 2; ++mt)
#pragma unroll
      for (int nt = 0; nt < 2; ++nt) {
        accR[mt * 2 + nt] = wmma_bf16(a[mt], br[nt], accR[mt * 2 + nt]);
        accI[mt * 2 + nt] = wmma_bf16(a[mt], bi[nt], accI[mt * 2 + nt]);
      }
  }

  // Epilogue: per-element complex multiply by wfilt(t, hw, bin), store bf16.
  int rlo = lane & 15;
  int hi8 = (lane >> 4) * 8;
#pragma unroll
  for (int mt = 0; mt < 2; ++mt)
#pragma unroll
    for (int nt = 0; nt < 2; ++nt) {
      v8f r  = accR[mt * 2 + nt];
      v8f im = accI[mt * 2 + nt];
#pragma unroll
      for (int v = 0; v < 8; ++v) {
        int m   = m0 + wm * 32 + mt * 16 + v + hi8;
        int bin = n0 + wn * 32 + nt * 16 + rlo;
        int bt  = m / HWDIM;
        int hw  = m - bt * HWDIM;
        int t   = bt & 7;                       // bt = b*T + t, T=8
        float wr = 0.0f, wi = 0.0f;
        if (bin < 513) {
          const float* wp = wfilt + (((size_t)(t * HWDIM + hw)) * 513 + bin) * 2;
          wr = wp[0]; wi = wp[1];
        }
        float re = r[v], iv = im[v];
        size_t o = (size_t)m * NBIN + bin;
        Re[o] = (__bf16)(re * wr - iv * wi);
        Im[o] = (__bf16)(re * wi + iv * wr);
      }
    }
}

// ---------------- GEMM 2: inverse rDFT ----------------
// Dm[m, c] = sum_bin Re[m,bin]*Brt[c,bin] + Im[m,bin]*Bit[c,bin]

__global__ __launch_bounds__(128)
void gemm_inv_fft_kernel(const __bf16* __restrict__ Re, const __bf16* __restrict__ Im,
                         const __bf16* __restrict__ Brt, const __bf16* __restrict__ Bit,
                         __bf16* __restrict__ Dm) {
  __shared__ __bf16 sAr[2][64 * LDA];
  __shared__ __bf16 sAi[2][64 * LDA];
  __shared__ __bf16 sBr[2][64 * LDA];
  __shared__ __bf16 sBi[2][64 * LDA];
  int tid  = threadIdx.x;
  int lane = tid & 31;
  int wave = tid >> 5;
  int wm = wave >> 1, wn = wave & 1;
  int m0 = blockIdx.x * 64;
  int n0 = blockIdx.y * 64;

  // Per-wave TDM job: one full 64-row tile each.
  const __bf16* gbase;
  __bf16 *l0, *l1;
  if (wave == 0)      { gbase = Re  + (size_t)m0 * NBIN; l0 = sAr[0]; l1 = sAr[1]; }
  else if (wave == 1) { gbase = Im  + (size_t)m0 * NBIN; l0 = sAi[0]; l1 = sAi[1]; }
  else if (wave == 2) { gbase = Brt + (size_t)n0 * NBIN; l0 = sBr[0]; l1 = sBr[1]; }
  else                { gbase = Bit + (size_t)n0 * NBIN; l0 = sBi[0]; l1 = sBi[1]; }

  tdm_load_tile(l0, gbase, 64, NBIN);

  v8f acc[4] = {};

  for (int k0 = 0; k0 < NBIN; k0 += 32) {
    int cur = (k0 >> 5) & 1;
    wait_tensor0();
    __syncthreads();
    if (k0 + 32 < NBIN)
      tdm_load_tile(cur ? l0 : l1, gbase + k0 + 32, 64, NBIN);

    const __bf16* arT = sAr[cur];
    const __bf16* aiT = sAi[cur];
    const __bf16* brT = sBr[cur];
    const __bf16* biT = sBi[cur];
    v16bf ar[2], ai[2], br[2], bi[2];
#pragma unroll
    for (int mt = 0; mt < 2; ++mt) { ar[mt] = lds_frag(arT + (wm * 32 + mt * 16) * LDA, lane);
                                     ai[mt] = lds_frag(aiT + (wm * 32 + mt * 16) * LDA, lane); }
#pragma unroll
    for (int nt = 0; nt < 2; ++nt) { br[nt] = lds_frag(brT + (wn * 32 + nt * 16) * LDA, lane);
                                     bi[nt] = lds_frag(biT + (wn * 32 + nt * 16) * LDA, lane); }
#pragma unroll
    for (int mt = 0; mt < 2; ++mt)
#pragma unroll
      for (int nt = 0; nt < 2; ++nt) {
        acc[mt * 2 + nt] = wmma_bf16(ar[mt], br[nt], acc[mt * 2 + nt]);
        acc[mt * 2 + nt] = wmma_bf16(ai[mt], bi[nt], acc[mt * 2 + nt]);
      }
  }

  int rlo = lane & 15;
  int hi8 = (lane >> 4) * 8;
#pragma unroll
  for (int mt = 0; mt < 2; ++mt)
#pragma unroll
    for (int nt = 0; nt < 2; ++nt) {
      v8f a = acc[mt * 2 + nt];
#pragma unroll
      for (int v = 0; v < 8; ++v) {
        int m = m0 + wm * 32 + mt * 16 + v + hi8;
        int n = n0 + wn * 32 + nt * 16 + rlo;
        Dm[(size_t)m * CDIM + n] = (__bf16)a[v];
      }
    }
}

// ---------------- GEMM 3: 1x1 conv + bias + exact GELU + residual ----------------
// out[bt, o, hw] = x[bt, o, hw] + gelu(sum_c Dm[m,c]*conv_w[o,c] + bias[o])

__global__ __launch_bounds__(128)
void gemm_conv_gelu_kernel(const __bf16* __restrict__ Dm, const __bf16* __restrict__ Wb,
                           const float* __restrict__ bias, const float* __restrict__ x,
                           float* __restrict__ out) {
  __shared__ __bf16 sA[2][64 * LDA];
  __shared__ __bf16 sB[2][64 * LDA];
  int tid  = threadIdx.x;
  int lane = tid & 31;
  int wave = tid >> 5;
  int wm = wave >> 1, wn = wave & 1;
  int m0 = blockIdx.x * 64;
  int n0 = blockIdx.y * 64;

  // Per-wave TDM job: A and B tiles each split lo/hi across two waves.
  const __bf16* gbase;
  __bf16 *l0, *l1;
  if (wave == 0)      { gbase = Dm + (size_t)m0 * CDIM;        l0 = sA[0];            l1 = sA[1]; }
  else if (wave == 1) { gbase = Dm + (size_t)(m0 + 32) * CDIM; l0 = sA[0] + 32 * LDA; l1 = sA[1] + 32 * LDA; }
  else if (wave == 2) { gbase = Wb + (size_t)n0 * CDIM;        l0 = sB[0];            l1 = sB[1]; }
  else                { gbase = Wb + (size_t)(n0 + 32) * CDIM; l0 = sB[0] + 32 * LDA; l1 = sB[1] + 32 * LDA; }

  tdm_load_tile(l0, gbase, 32, CDIM);

  v8f acc[4] = {};

  for (int k0 = 0; k0 < CDIM; k0 += 32) {
    int cur = (k0 >> 5) & 1;
    wait_tensor0();
    __syncthreads();
    if (k0 + 32 < CDIM)
      tdm_load_tile(cur ? l0 : l1, gbase + k0 + 32, 32, CDIM);

    const __bf16* aT = sA[cur];
    const __bf16* bT = sB[cur];
    v16bf a[2], b[2];
#pragma unroll
    for (int mt = 0; mt < 2; ++mt) a[mt] = lds_frag(aT + (wm * 32 + mt * 16) * LDA, lane);
#pragma unroll
    for (int nt = 0; nt < 2; ++nt) b[nt] = lds_frag(bT + (wn * 32 + nt * 16) * LDA, lane);
#pragma unroll
    for (int mt = 0; mt < 2; ++mt)
#pragma unroll
      for (int nt = 0; nt < 2; ++nt)
        acc[mt * 2 + nt] = wmma_bf16(a[mt], b[nt], acc[mt * 2 + nt]);
  }

  int rlo = lane & 15;
  int hi8 = (lane >> 4) * 8;
#pragma unroll
  for (int mt = 0; mt < 2; ++mt)
#pragma unroll
    for (int nt = 0; nt < 2; ++nt) {
      v8f a = acc[mt * 2 + nt];
#pragma unroll
      for (int v = 0; v < 8; ++v) {
        int m = m0 + wm * 32 + mt * 16 + v + hi8;
        int n = n0 + wn * 32 + nt * 16 + rlo;
        int bt = m / HWDIM;
        int hw = m - bt * HWDIM;
        float e = a[v] + bias[n];
        float g = 0.5f * e * (1.0f + erff(e * 0.70710678118654752f));  // exact GELU
        size_t idx = ((size_t)bt * CDIM + n) * HWDIM + hw;             // [bt, c, h, w]
        out[idx] = x[idx] + g;
      }
    }
}

// ---------------------------------------------------------------------------

extern "C" void kernel_launch(void* const* d_in, const int* in_sizes, int n_in,
                              void* d_out, int out_size, void* d_ws, size_t ws_size,
                              hipStream_t stream) {
  const float* x      = (const float*)d_in[0];   // [128, 1024, 14, 14]
  const float* wfilt  = (const float*)d_in[1];   // [8, 14, 14, 513, 2]
  const float* conv_w = (const float*)d_in[2];   // [1024, 1024]
  const float* conv_b = (const float*)d_in[3];   // [1024]
  float* out = (float*)d_out;

  // Workspace carve-up (256B aligned slabs).
  char* ws = (char*)d_ws;
  size_t off = 0;
  auto carve = [&](size_t bytes) -> char* {
    char* p = ws + off;
    off = (off + bytes + 255) & ~(size_t)255;
    return p;
  };
  __bf16* Xc  = (__bf16*)carve((size_t)M_ROWS * CDIM * sizeof(__bf16));
  __bf16* Re  = (__bf16*)carve((size_t)M_ROWS * NBIN * sizeof(__bf16));
  __bf16* Im  = (__bf16*)carve((size_t)M_ROWS * NBIN * sizeof(__bf16));
  __bf16* Dm  = (__bf16*)carve((size_t)M_ROWS * CDIM * sizeof(__bf16));
  __bf16* Frt = (__bf16*)carve((size_t)NBIN * CDIM * sizeof(__bf16));
  __bf16* Fit = (__bf16*)carve((size_t)NBIN * CDIM * sizeof(__bf16));
  __bf16* Brt = (__bf16*)carve((size_t)CDIM * NBIN * sizeof(__bf16));
  __bf16* Bit = (__bf16*)carve((size_t)CDIM * NBIN * sizeof(__bf16));
  __bf16* Wb  = (__bf16*)carve((size_t)CDIM * CDIM * sizeof(__bf16));
  (void)ws_size; (void)in_sizes; (void)n_in; (void)out_size;

  // Setup: DFT bases, bf16 weights, channels-last transpose.
  init_basis_kernel<<<(NBIN * CDIM + 255) / 256, 256, 0, stream>>>(Frt, Fit, Brt, Bit);
  convert_w_kernel<<<(CDIM * CDIM + 255) / 256, 256, 0, stream>>>(conv_w, Wb);
  transpose_cl_kernel<<<dim3(128, 7, 32), dim3(32, 8), 0, stream>>>(x, Xc);

  // GEMM 1: forward rDFT (Re & Im) + spectral filter.   M/64=392, NBIN/64=9
  gemm_fwd_fft_kernel<<<dim3(392, 9), 128, 0, stream>>>(Xc, Frt, Fit, wfilt, Re, Im);

  // GEMM 2: inverse rDFT.                               M/64=392, CDIM/64=16
  gemm_inv_fft_kernel<<<dim3(392, 16), 128, 0, stream>>>(Re, Im, Brt, Bit, Dm);

  // GEMM 3: 1x1 conv + bias + GELU + residual, scatter to [bt, c, h, w].
  gemm_conv_gelu_kernel<<<dim3(392, 16), 128, 0, stream>>>(Dm, Wb, conv_b, x, out);
}
